// TransformerBlock_38491496907080
// MI455X (gfx1250) — matrix operand order repaired and checked by reading, not compiled, not run on hardware
//
#include <hip/hip_runtime.h>
#include <hip/hip_bf16.h>
#include <math.h>

// ---------------- problem constants ----------------
#define BB   4
#define TT   2048
#define DD   1024
#define HH   16
#define HDIM 64
#define FFD  4096
#define MROWS (BB * TT)   // 8192

// ---------------- types ----------------
typedef __bf16 v16bf __attribute__((ext_vector_type(16)));
typedef float  v8f   __attribute__((ext_vector_type(8)));
typedef unsigned int ui4 __attribute__((ext_vector_type(4)));

union F16Frag {
    v16bf v;
    ui4   q[2];
    unsigned short s[16];
};

__device__ __forceinline__ unsigned short f2bf(float f) {
    unsigned int u = __float_as_uint(f);
    unsigned int r = u + 0x7FFFu + ((u >> 16) & 1u);   // round-to-nearest-even
    return (unsigned short)(r >> 16);
}

// async copy 16B global -> LDS (per lane), tracked by ASYNCcnt
__device__ __forceinline__ void async_cp16(const void* gptr, void* lptr) {
    unsigned lds = (unsigned)(uintptr_t)lptr;  // low 32 bits = LDS offset
    unsigned long long ga = (unsigned long long)(uintptr_t)gptr;
    asm volatile("global_load_async_to_lds_b128 %0, %1, off"
                 :: "v"(lds), "v"(ga) : "memory");
}
__device__ __forceinline__ void async_wait0() {
    asm volatile("s_wait_asynccnt 0x0" ::: "memory");
}

// ---------------- fp32 -> bf16 transpose (weights) ----------------
// in: [K][N] fp32 row-major; out: [N][K] bf16 row-major
__global__ __launch_bounds__(256)
void cvt_t_kernel(const float* __restrict__ in, unsigned short* __restrict__ out,
                  int K, int N) {
    __shared__ float tile[32][33];
    const int n0 = blockIdx.x * 32;
    const int k0 = blockIdx.y * 32;
    const int tx = threadIdx.x & 31;
    const int ty = threadIdx.x >> 5;   // 0..7
#pragma unroll
    for (int i = 0; i < 4; ++i) {
        const int k = k0 + ty + i * 8;
        tile[ty + i * 8][tx] = in[(size_t)k * N + n0 + tx];
    }
    __syncthreads();
#pragma unroll
    for (int i = 0; i < 4; ++i) {
        const int n = n0 + ty + i * 8;
        out[(size_t)n * K + k0 + tx] = f2bf(tile[tx][ty + i * 8]);
    }
}

// ---------------- LayerNorm (fp32 in -> bf16 out) ----------------
__global__ __launch_bounds__(256)
void ln_kernel(const float* __restrict__ x, const float* __restrict__ g,
               const float* __restrict__ be, unsigned short* __restrict__ out) {
    __shared__ float s1[256];
    __shared__ float s2[256];
    const int row = blockIdx.x;
    const int t = threadIdx.x;
    const float* xr = x + (size_t)row * DD;
    float4 v = ((const float4*)xr)[t];
    float s = v.x + v.y + v.z + v.w;
    float q = v.x*v.x + v.y*v.y + v.z*v.z + v.w*v.w;
    s1[t] = s; s2[t] = q;
    __syncthreads();
    for (int off = 128; off > 0; off >>= 1) {
        if (t < off) { s1[t] += s1[t + off]; s2[t] += s2[t + off]; }
        __syncthreads();
    }
    const float mean = s1[0] * (1.0f / DD);
    const float var  = s2[0] * (1.0f / DD) - mean * mean;
    const float rstd = rsqrtf(var + 1e-5f);
    float4 gv = ((const float4*)g)[t];
    float4 bv = ((const float4*)be)[t];
    unsigned short* orow = out + (size_t)row * DD + t * 4;
    orow[0] = f2bf((v.x - mean) * rstd * gv.x + bv.x);
    orow[1] = f2bf((v.y - mean) * rstd * gv.y + bv.y);
    orow[2] = f2bf((v.z - mean) * rstd * gv.z + bv.z);
    orow[3] = f2bf((v.w - mean) * rstd * gv.w + bv.w);
}

// ---------------- WMMA GEMM (double-buffered, async LDS staging) ----------------
// C[M,N] = A[M,K] (bf16 rowmaj) @ B (given PRE-TRANSPOSED: Bt[N][K] bf16 rowmaj)
// MODE 0: store bf16
// MODE 1: +bias, exact GELU, store bf16
// MODE 2: +bias, +residual (fp32), store fp32
// MODE 3: qkv: q/k cols -> bf16 rowmajor, v cols -> Vt[b][h][hd][t] bf16
template <int MODE>
__global__ __launch_bounds__(256)
void gemm_bf16_kernel(const unsigned short* __restrict__ A,
                      const unsigned short* __restrict__ Bt,
                      int M, int N, int K,
                      const float* __restrict__ bias,
                      const float* __restrict__ res,
                      float* __restrict__ Cf,
                      unsigned short* __restrict__ Cb,
                      unsigned short* __restrict__ Vt) {
    __shared__ __align__(16) unsigned short sA[2][128][32];    // 16 KB
    __shared__ __align__(16) unsigned short sBT[2][256][32];   // 32 KB, [n][k]

    const int tid  = threadIdx.x;
    const int lane = tid & 31;
    const int w    = tid >> 5;       // 8 waves
    const int wm   = w >> 2;         // 0..1
    const int wn   = w & 3;          // 0..3
    const int half = lane >> 4;
    const int l16  = lane & 15;
    const int bm   = blockIdx.y * 128;
    const int bn   = blockIdx.x * 256;

    v8f acc[4][4];
#pragma unroll
    for (int mi = 0; mi < 4; ++mi)
#pragma unroll
        for (int ni = 0; ni < 4; ++ni)
#pragma unroll
            for (int j = 0; j < 8; ++j) acc[mi][ni][j] = 0.0f;

    // async prefetch of one K-chunk into buffer `buf`
    auto prefetch = [&](int buf, int kb) {
        {   // A tile: 128 rows x 32 k (64B/row); 2 x 16B per thread
            const int row = tid >> 1;
            const int seg = (tid & 1) * 16;
            const unsigned short* ap = A + (size_t)(bm + row) * K + kb + seg;
            async_cp16(ap,     &sA[buf][row][seg]);
            async_cp16(ap + 8, &sA[buf][row][seg + 8]);
        }
        {   // B tile (pre-transposed): 256 n-rows x 32 k; 4 x 16B per thread
            const unsigned short* bp = Bt + (size_t)(bn + tid) * K + kb;
            async_cp16(bp,      &sBT[buf][tid][0]);
            async_cp16(bp + 8,  &sBT[buf][tid][8]);
            async_cp16(bp + 16, &sBT[buf][tid][16]);
            async_cp16(bp + 24, &sBT[buf][tid][24]);
        }
    };

    prefetch(0, 0);

    for (int kb = 0; kb < K; kb += 32) {
        const int cur = (kb >> 5) & 1;
        async_wait0();        // my copies into `cur` (and earlier) done
        __syncthreads();      // everyone's copies done; everyone done reading cur^1
        if (kb + 32 < K) prefetch(cur ^ 1, kb + 32);

        F16Frag Af[4], Bf[4];
#pragma unroll
        for (int mi = 0; mi < 4; ++mi) {
            const int row = wm * 64 + mi * 16 + l16;
            Af[mi].q[0] = *(const ui4*)&sA[cur][row][half * 8];
            Af[mi].q[1] = *(const ui4*)&sA[cur][row][16 + half * 8];
        }
#pragma unroll
        for (int ni = 0; ni < 4; ++ni) {
            const int col = wn * 64 + ni * 16 + l16;
            Bf[ni].q[0] = *(const ui4*)&sBT[cur][col][half * 8];
            Bf[ni].q[1] = *(const ui4*)&sBT[cur][col][16 + half * 8];
        }
#pragma unroll
        for (int mi = 0; mi < 4; ++mi)
#pragma unroll
            for (int ni = 0; ni < 4; ++ni)
                acc[mi][ni] = __builtin_amdgcn_wmma_f32_16x16x32_bf16(
                    false, Af[mi].v, false, Bf[ni].v, (short)0, acc[mi][ni],
                    false, false);
    }

    // epilogue
#pragma unroll
    for (int mi = 0; mi < 4; ++mi)
#pragma unroll
        for (int ni = 0; ni < 4; ++ni) {
            const int gc = bn + wn * 64 + ni * 16 + l16;
#pragma unroll
            for (int j = 0; j < 8; ++j) {
                const int gr = bm + wm * 64 + mi * 16 + j + 8 * half;
                const size_t idx = (size_t)gr * N + gc;
                float v = acc[mi][ni][j];
                if (MODE == 0) {
                    Cb[idx] = f2bf(v);
                } else if (MODE == 1) {
                    v += bias[gc];
                    v = 0.5f * v * (1.0f + erff(v * 0.70710678118654752f));
                    Cb[idx] = f2bf(v);
                } else if (MODE == 2) {
                    v += bias[gc] + res[idx];
                    Cf[idx] = v;
                } else {
                    // MODE 3: qkv projection
                    if (gc < 2 * DD) {
                        Cb[idx] = f2bf(v);     // q or k, row-major
                    } else {
                        const int bidx = gr >> 11;        // / TT
                        const int t    = gr & (TT - 1);
                        const int c    = gc - 2 * DD;
                        const int hh   = c >> 6;
                        const int hd   = c & 63;
                        Vt[(((size_t)bidx * HH + hh) * HDIM + hd) * TT + t] = f2bf(v);
                    }
                }
            }
        }
}

// ---------------- flash attention (causal) ----------------
// qkv: bf16 [B, T, 3*D] (q at col h*64, k at D + h*64; v region unused)
// vt : bf16 [B][H][HD][T]
// o  : bf16 [B, T, D]
__global__ __launch_bounds__(128)
void attn_kernel(const unsigned short* __restrict__ qkv,
                 const unsigned short* __restrict__ vt,
                 unsigned short* __restrict__ o) {
    __shared__ __align__(16) unsigned short sK[32][64];      // [key][dim]
    __shared__ __align__(16) unsigned short sVT[64][32];     // [dim][key]
    __shared__ __align__(16) unsigned short sP[4][16][32];   // per-wave P tile

    const int tid  = threadIdx.x;
    const int lane = tid & 31;
    const int w    = tid >> 5;       // wave 0..3 -> q rows w*16..+15
    const int half = lane >> 4;
    const int l16  = lane & 15;

    int blk = blockIdx.x;
    const int qtiles = TT / 64;
    const int qb = blk % qtiles; blk /= qtiles;
    const int h  = blk % HH;     blk /= HH;
    const int b  = blk;
    const int qbase = qb * 64;

    const size_t rowstride = 3 * DD;
    const size_t baseBT = (size_t)b * TT;
    const unsigned short* vthead = vt + ((size_t)b * HH + h) * HDIM * TT;

    // preload Q A-fragments (dims 0-31, 32-63)
    F16Frag Qf[2];
    {
        const int qrow = qbase + w * 16 + l16;
        const unsigned short* qp = qkv + (baseBT + qrow) * rowstride + h * HDIM;
#pragma unroll
        for (int c = 0; c < 2; ++c) {
            const int kc = c * 32;
            Qf[c].q[0] = *(const ui4*)(qp + kc + half * 8);
            Qf[c].q[1] = *(const ui4*)(qp + kc + 16 + half * 8);
        }
    }

    v8f Oacc[4];
#pragma unroll
    for (int n = 0; n < 4; ++n)
#pragma unroll
        for (int j = 0; j < 8; ++j) Oacc[n][j] = 0.0f;

    float rmax[8], rsum[8];
#pragma unroll
    for (int j = 0; j < 8; ++j) { rmax[j] = -INFINITY; rsum[j] = 0.0f; }

    const float scale = 0.125f;               // 1/sqrt(64)
    const int nchunks = (qbase + 64) / 32;    // causal: keys up to block max q

    for (int c = 0; c < nchunks; ++c) {
        const int kbase = c * 32;
        // async load K chunk (32 keys x 64 dims, row-major): 2 x 16B / thread
        {
            const int key = tid >> 2;
            const int seg = (tid & 3) * 16;
            const unsigned short* kp =
                qkv + (baseBT + kbase + key) * rowstride + DD + h * HDIM + seg;
            async_cp16(kp,     &sK[key][seg]);
            async_cp16(kp + 8, &sK[key][seg + 8]);
        }
        // async load V^T chunk ([dim][key], linear from vt): 2 x 16B / thread
        {
            const int dim = tid >> 1;
            const int seg = (tid & 1) * 16;
            const unsigned short* vp = vthead + (size_t)dim * TT + kbase + seg;
            async_cp16(vp,     &sVT[dim][seg]);
            async_cp16(vp + 8, &sVT[dim][seg + 8]);
        }
        async_wait0();
        __syncthreads();

        // scores: S0 = keys kbase..+15, S1 = keys kbase+16..+31
        v8f S0, S1;
#pragma unroll
        for (int j = 0; j < 8; ++j) { S0[j] = 0.0f; S1[j] = 0.0f; }
        {
            F16Frag Bf;
            // kt = 0
            Bf.q[0] = *(const ui4*)&sK[l16][half * 8];
            Bf.q[1] = *(const ui4*)&sK[l16][16 + half * 8];
            S0 = __builtin_amdgcn_wmma_f32_16x16x32_bf16(false, Qf[0].v, false, Bf.v, (short)0, S0, false, false);
            Bf.q[0] = *(const ui4*)&sK[l16][32 + half * 8];
            Bf.q[1] = *(const ui4*)&sK[l16][32 + 16 + half * 8];
            S0 = __builtin_amdgcn_wmma_f32_16x16x32_bf16(false, Qf[1].v, false, Bf.v, (short)0, S0, false, false);
            // kt = 1
            Bf.q[0] = *(const ui4*)&sK[16 + l16][half * 8];
            Bf.q[1] = *(const ui4*)&sK[16 + l16][16 + half * 8];
            S1 = __builtin_amdgcn_wmma_f32_16x16x32_bf16(false, Qf[0].v, false, Bf.v, (short)0, S1, false, false);
            Bf.q[0] = *(const ui4*)&sK[16 + l16][32 + half * 8];
            Bf.q[1] = *(const ui4*)&sK[16 + l16][32 + 16 + half * 8];
            S1 = __builtin_amdgcn_wmma_f32_16x16x32_bf16(false, Qf[1].v, false, Bf.v, (short)0, S1, false, false);
        }

        // online softmax over this 32-key block
#pragma unroll
        for (int j = 0; j < 8; ++j) {
            const int qrow = qbase + w * 16 + j + 8 * half;
            const int k0 = kbase + l16;
            const int k1 = kbase + 16 + l16;
            float a0 = (k0 <= qrow) ? S0[j] * scale : -INFINITY;
            float a1 = (k1 <= qrow) ? S1[j] * scale : -INFINITY;
            float m = fmaxf(a0, a1);
#pragma unroll
            for (int off = 8; off > 0; off >>= 1)
                m = fmaxf(m, __shfl_xor(m, off, 32));
            const float mn = fmaxf(rmax[j], m);
            const float alpha = __expf(rmax[j] - mn);
            rmax[j] = mn;
            const float p0 = __expf(a0 - mn);
            const float p1 = __expf(a1 - mn);
            float ps = p0 + p1;
#pragma unroll
            for (int off = 8; off > 0; off >>= 1)
                ps += __shfl_xor(ps, off, 32);
            rsum[j] = rsum[j] * alpha + ps;
#pragma unroll
            for (int n = 0; n < 4; ++n) Oacc[n][j] *= alpha;
            const int row = j + 8 * half;
            sP[w][row][l16]      = f2bf(p0);
            sP[w][row][16 + l16] = f2bf(p1);
        }
        __syncthreads();

        // P @ V
        F16Frag Pf;
        Pf.q[0] = *(const ui4*)&sP[w][l16][half * 8];
        Pf.q[1] = *(const ui4*)&sP[w][l16][16 + half * 8];
#pragma unroll
        for (int n = 0; n < 4; ++n) {
            F16Frag Vf;
            const int col = n * 16 + l16;
            Vf.q[0] = *(const ui4*)&sVT[col][half * 8];
            Vf.q[1] = *(const ui4*)&sVT[col][16 + half * 8];
            Oacc[n] = __builtin_amdgcn_wmma_f32_16x16x32_bf16(
                false, Pf.v, false, Vf.v, (short)0, Oacc[n], false, false);
        }
        __syncthreads();
    }

    // normalize and store o[b, qrow, h*HDIM + hd] as bf16
#pragma unroll
    for (int n = 0; n < 4; ++n)
#pragma unroll
        for (int j = 0; j < 8; ++j) {
            const int qrow = qbase + w * 16 + j + 8 * half;
            const size_t idx = (baseBT + qrow) * DD + h * HDIM + n * 16 + l16;
            o[idx] = f2bf(Oacc[n][j] / rsum[j]);
        }
}

// ---------------- host launcher ----------------
extern "C" void kernel_launch(void* const* d_in, const int* in_sizes, int n_in,
                              void* d_out, int out_size, void* d_ws, size_t ws_size,
                              hipStream_t stream) {
    const float* x     = (const float*)d_in[0];
    // d_in[1] = mask (causality implemented directly)
    const float* ln1_g = (const float*)d_in[2];
    const float* ln1_b = (const float*)d_in[3];
    const float* ln2_g = (const float*)d_in[4];
    const float* ln2_b = (const float*)d_in[5];
    const float* qkv_w = (const float*)d_in[6];
    const float* out_w = (const float*)d_in[7];
    const float* out_b = (const float*)d_in[8];
    const float* fc1_w = (const float*)d_in[9];
    const float* fc1_b = (const float*)d_in[10];
    const float* fc2_w = (const float*)d_in[11];
    const float* fc2_b = (const float*)d_in[12];

    // workspace carve (256B aligned)
    char* p = (char*)d_ws;
    auto carve = [&](size_t bytes) {
        void* r = (void*)p;
        p += (bytes + 255) & ~(size_t)255;
        return r;
    };
    unsigned short* h_bf     = (unsigned short*)carve((size_t)MROWS * DD * 2);
    unsigned short* qkvw_t   = (unsigned short*)carve((size_t)DD * 3 * DD * 2);   // [3D][D]
    unsigned short* outw_t   = (unsigned short*)carve((size_t)DD * DD * 2);       // [D][D]
    unsigned short* fc1w_t   = (unsigned short*)carve((size_t)DD * FFD * 2);      // [FF][D]
    unsigned short* fc2w_t   = (unsigned short*)carve((size_t)FFD * DD * 2);      // [D][FF]
    unsigned short* qkv_bf   = (unsigned short*)carve((size_t)MROWS * 3 * DD * 2);
    unsigned short* vt_bf    = (unsigned short*)carve((size_t)BB * HH * HDIM * TT * 2);
    unsigned short* o_bf     = (unsigned short*)carve((size_t)MROWS * DD * 2);
    float*          x1       = (float*)carve((size_t)MROWS * DD * 4);
    unsigned short* h2_bf    = (unsigned short*)carve((size_t)MROWS * DD * 2);
    unsigned short* ff_bf    = (unsigned short*)carve((size_t)MROWS * FFD * 2);

    // weight convert + transpose: W[K][N] fp32 -> Wt[N][K] bf16
    cvt_t_kernel<<<dim3(3 * DD / 32, DD / 32), 256, 0, stream>>>(qkv_w, qkvw_t, DD, 3 * DD);
    cvt_t_kernel<<<dim3(DD / 32, DD / 32),     256, 0, stream>>>(out_w, outw_t, DD, DD);
    cvt_t_kernel<<<dim3(FFD / 32, DD / 32),    256, 0, stream>>>(fc1_w, fc1w_t, DD, FFD);
    cvt_t_kernel<<<dim3(DD / 32, FFD / 32),    256, 0, stream>>>(fc2_w, fc2w_t, FFD, DD);

    // LN1
    ln_kernel<<<MROWS, 256, 0, stream>>>(x, ln1_g, ln1_b, h_bf);

    // QKV projection: q,k row-major bf16; v written transposed to vt_bf
    gemm_bf16_kernel<3><<<dim3(3 * DD / 256, MROWS / 128), 256, 0, stream>>>(
        h_bf, qkvw_t, MROWS, 3 * DD, DD, nullptr, nullptr, nullptr, qkv_bf, vt_bf);

    // attention
    attn_kernel<<<BB * HH * (TT / 64), 128, 0, stream>>>(qkv_bf, vt_bf, o_bf);

    // out projection + bias + residual -> fp32 x1
    gemm_bf16_kernel<2><<<dim3(DD / 256, MROWS / 128), 256, 0, stream>>>(
        o_bf, outw_t, MROWS, DD, DD, out_b, x, x1, nullptr, nullptr);

    // LN2
    ln_kernel<<<MROWS, 256, 0, stream>>>(x1, ln2_g, ln2_b, h2_bf);

    // FC1 + bias + exact GELU -> bf16
    gemm_bf16_kernel<1><<<dim3(FFD / 256, MROWS / 128), 256, 0, stream>>>(
        h2_bf, fc1w_t, MROWS, FFD, DD, fc1_b, nullptr, nullptr, ff_bf, nullptr);

    // FC2 + bias + residual -> fp32 output
    gemm_bf16_kernel<2><<<dim3(DD / 256, MROWS / 128), 256, 0, stream>>>(
        ff_bf, fc2w_t, MROWS, DD, FFD, fc2_b, x1, (float*)d_out, nullptr, nullptr);
}
